// MultiHeadedSelfAttention_29910152249658
// MI455X (gfx1250) — matrix operand order repaired and checked by reading, not compiled, not run on hardware
//
#include <hip/hip_runtime.h>
#include <hip/hip_bf16.h>

typedef _Float16 v16h __attribute__((ext_vector_type(16)));
typedef _Float16 v8h  __attribute__((ext_vector_type(8)));
typedef float    v8f  __attribute__((ext_vector_type(8)));
typedef unsigned int uint32x4 __attribute__((ext_vector_type(4)));
typedef int          int32x8  __attribute__((ext_vector_type(8)));
typedef int          int32x4  __attribute__((ext_vector_type(4)));

#define NB     2
#define TSEQ   2048
#define DMODEL 1024
#define NH     16
#define HD     64

#define CAT16(lo, hi) __builtin_shufflevector(lo, hi, 0, 1, 2, 3, 4, 5, 6, 7, \
                                              8, 9, 10, 11, 12, 13, 14, 15)

// ---------------------------------------------------------------------------
// Kernel 0: one-time data prep. x -> f16 (contiguous), weights -> f16
// transposed [tensor][h][n][d] so GEMM B-fragments are contiguous b128 loads.
// ---------------------------------------------------------------------------
__global__ __launch_bounds__(256) void prep_kernel(
    const float* __restrict__ x,
    const float* __restrict__ Wq, const float* __restrict__ Wk,
    const float* __restrict__ Wv,
    _Float16* __restrict__ xh, _Float16* __restrict__ Wt)
{
    const int tid    = blockIdx.x * 256 + threadIdx.x;
    const int stride = gridDim.x * 256;
    for (int i = tid; i < NB * TSEQ * DMODEL; i += stride)
        xh[i] = (_Float16)x[i];
    // Wt[((t*NH+h)*HD + n)*DMODEL + d] = W[t][h][d][n]
    for (int o = tid; o < 3 * NH * HD * DMODEL; o += stride) {
        const int d = o & (DMODEL - 1);
        const int n = (o >> 10) & (HD - 1);
        const int h = (o >> 16) & (NH - 1);
        const int t = o >> 20;
        const float* W = (t == 0) ? Wq : (t == 1) ? Wk : Wv;
        Wt[o] = (_Float16)W[h * (DMODEL * HD) + d * HD + n];
    }
}

// ---------------------------------------------------------------------------
// Kernel 1: fused QKV projection GEMM (f16 WMMA, f32 accumulate).
// One wave = 16 rows x 64 cols (one head) of one of Q/K/V.
// Q pre-scaled by 1/sqrt(1024); V stored transposed [B,H,64,T].
// ---------------------------------------------------------------------------
__global__ __launch_bounds__(256) void qkv_proj_kernel(
    const _Float16* __restrict__ xh, const _Float16* __restrict__ Wt,
    const float* __restrict__ bq, const float* __restrict__ bk,
    const float* __restrict__ bv,
    _Float16* __restrict__ Qb, _Float16* __restrict__ Kb,
    _Float16* __restrict__ Vt)
{
    const int lane  = threadIdx.x & 31;
    const int gwave = blockIdx.x * 8 + (threadIdx.x >> 5);   // 0..12287
    const int rt     = gwave & 255;
    const int th     = gwave >> 8;
    const int tensor = th / NH;
    const int h      = th - tensor * NH;

    const float* bs = (tensor == 0) ? bq : (tensor == 1) ? bk : bv;

    const int lo  = lane & 15;
    const int hi  = lane >> 4;
    const int kb8 = hi * 8;

    const _Float16* xrow  = xh + (size_t)(rt * 16 + lo) * DMODEL;
    const _Float16* wbase = Wt + (size_t)(tensor * NH + h) * HD * DMODEL;

    v8f acc[4] = {{}, {}, {}, {}};

    for (int k0 = 0; k0 < DMODEL; k0 += 32) {
        const v8h alo = *(const v8h*)(xrow + k0 + kb8);
        const v8h ahi = *(const v8h*)(xrow + k0 + kb8 + 16);
        const v16h a = CAT16(alo, ahi);
#pragma unroll
        for (int c = 0; c < 4; ++c) {
            const _Float16* wp =
                wbase + (size_t)(c * 16 + lo) * DMODEL + k0 + hi * 8;
            const v8h b0 = *(const v8h*)wp;
            const v8h b1 = *(const v8h*)(wp + 16);
            const v16h bf = CAT16(b0, b1);
            acc[c] = __builtin_amdgcn_wmma_f32_16x16x32_f16(
                false, a, false, bf, (short)0, acc[c], false, false);
        }
    }

#pragma unroll
    for (int c = 0; c < 4; ++c) {
        const int ncol = c * 16 + lo;
        const float bias = bs[h * HD + ncol];
#pragma unroll
        for (int r = 0; r < 8; ++r) {
            const int gr = rt * 16 + r + 8 * hi;
            const int bb = gr >> 11;
            const int t  = gr & (TSEQ - 1);
            float val = acc[c][r] + bias;
            if (tensor == 0) val *= 0.03125f;        // 1/sqrt(1024)
            const _Float16 hv = (_Float16)val;
            if (tensor == 0)
                Qb[((size_t)(bb * NH + h) * TSEQ + t) * HD + ncol] = hv;
            else if (tensor == 1)
                Kb[((size_t)(bb * NH + h) * TSEQ + t) * HD + ncol] = hv;
            else
                Vt[((size_t)(bb * NH + h) * HD + ncol) * TSEQ + t] = hv;
        }
    }
}

// ---------------------------------------------------------------------------
// TDM descriptor helpers (CDNA5 ISA ch.8): 2D tile, f16 elements.
// ---------------------------------------------------------------------------
__device__ __forceinline__ uint32x4 tdm_g0(unsigned lds_addr,
                                           unsigned long long gaddr) {
    uint32x4 g;
    g.x = 1u;                                       // count=1, user descriptor
    g.y = lds_addr;                                 // LDS byte address
    g.z = (unsigned)(gaddr & 0xffffffffull);        // global_addr[31:0]
    g.w = (unsigned)((gaddr >> 32) & 0x01ffffffull) // global_addr[56:32]
          | (2u << 30);                             // type=2 ("image")
    return g;
}

__device__ __forceinline__ int32x8 tdm_g1(unsigned tensor_d0, unsigned tensor_d1,
                                          unsigned tile_d0, unsigned tile_d1,
                                          unsigned long long stride0) {
    int32x8 g;
    g[0] = (int)(1u << 16);                          // data_size=1 (2 bytes)
    g[1] = (int)((tensor_d0 & 0xffffu) << 16);       // tensor_dim0[15:0]
    g[2] = (int)((tensor_d0 >> 16) | ((tensor_d1 & 0xffffu) << 16));
    g[3] = (int)((tensor_d1 >> 16) | (tile_d0 << 16));   // tile_dim0
    g[4] = (int)(tile_d1 & 0xffffu);                 // tile_dim1 (tile_dim2=0)
    g[5] = (int)(stride0 & 0xffffffffull);           // tensor_dim0_stride lo
    g[6] = (int)((stride0 >> 32) & 0xffffull);       // stride hi (dim1_stride=0)
    g[7] = 0;
    return g;
}

__device__ __forceinline__ void tdm_load(uint32x4 g0, int32x8 g1) {
    const int32x4 z4 = {0, 0, 0, 0};
    const int32x8 z8 = {0, 0, 0, 0, 0, 0, 0, 0};
    // amdgpu-toolchain (clang-23) 6-arg form: (g0, g1, g2, g3, g4?, cpol)
    __builtin_amdgcn_tensor_load_to_lds(g0, g1, z4, z4, z8, 0);
}

// ---------------------------------------------------------------------------
// Kernel 2: causal flash attention. 8 waves/block, one wave per 16 query rows.
// K/V tiles double-buffered in LDS by wave 0 via the Tensor Data Mover,
// overlapped with WMMA compute; TENSORcnt + workgroup barriers for sync.
// ---------------------------------------------------------------------------
__global__ __launch_bounds__(256) void attn_kernel(
    const _Float16* __restrict__ Qb, const _Float16* __restrict__ Kb,
    const _Float16* __restrict__ Vt, float* __restrict__ out)
{
    // [0,4096): K double buffer (2 x 32x64) | [4096,8192): V (2 x 64x32)
    // [8192,12288): per-wave P relayout staging (8 x 16x32)
    __shared__ _Float16 smem[12288];

    const int lane = threadIdx.x & 31;
    const int w    = threadIdx.x >> 5;
    const int idx  = blockIdx.x;            // 512 blocks
    const int rb   = idx & 15;
    const int h    = (idx >> 4) & 15;
    const int b    = idx >> 8;
    const int t0   = rb * 128 + w * 16;

    const _Float16* Qh = Qb + (size_t)(b * NH + h) * TSEQ * HD;
    const _Float16* Kh = Kb + (size_t)(b * NH + h) * TSEQ * HD;
    const _Float16* Vh = Vt + (size_t)(b * NH + h) * HD * TSEQ;

    const int lo  = lane & 15;
    const int hi  = lane >> 4;
    const int kb8 = hi * 8;

    // Q fragments: 16 rows x 64 features = two 16x32 A-frags in registers.
    v16h qa[2];
    {
        const _Float16* qrow = Qh + (size_t)(t0 + lo) * HD;
#pragma unroll
        for (int f = 0; f < 2; ++f) {
            const v8h q0 = *(const v8h*)(qrow + f * 32 + kb8);
            const v8h q1 = *(const v8h*)(qrow + f * 32 + kb8 + 16);
            qa[f] = CAT16(q0, q1);
        }
    }

    float m[8], l[8];
    v8f o[4] = {{}, {}, {}, {}};
#pragma unroll
    for (int r = 0; r < 8; ++r) { m[r] = -1e30f; l[r] = 0.0f; }

    const int nblk     = (t0 + 47) >> 5;            // this wave's causal range
    const int nblk_max = (rb * 128 + 159) >> 5;     // block-uniform loop bound

    const unsigned ldsbase = (unsigned)(unsigned long long)(uintptr_t)smem;

    // prologue: wave 0 issues TDM loads for block 0
    if (w == 0) {
        const unsigned long long kg =
            (unsigned long long)(uintptr_t)Kh;      // s0 = 0
        const unsigned long long vg = (unsigned long long)(uintptr_t)Vh;
        tdm_load(tdm_g0(ldsbase, kg), tdm_g1(HD, TSEQ, HD, 32, HD));
        tdm_load(tdm_g0(ldsbase + 8192u, vg), tdm_g1(TSEQ, HD, 32, HD, TSEQ));
    }

    for (int blk = 0; blk < nblk_max; ++blk) {
        if (w == 0) __builtin_amdgcn_s_wait_tensorcnt(0);
        __syncthreads();                            // tile for blk is visible

        if (w == 0 && blk + 1 < nblk_max) {         // prefetch blk+1 via TDM
            const int s1 = (blk + 1) * 32;
            const unsigned kbuf = ldsbase + ((unsigned)(blk + 1) & 1u) * 4096u;
            const unsigned vbuf = ldsbase + 8192u + ((unsigned)(blk + 1) & 1u) * 4096u;
            const unsigned long long kg =
                (unsigned long long)(uintptr_t)(Kh + (size_t)s1 * HD);
            const unsigned long long vg =
                (unsigned long long)(uintptr_t)(Vh + s1);
            tdm_load(tdm_g0(kbuf, kg), tdm_g1(HD, TSEQ, HD, 32, HD));
            tdm_load(tdm_g0(vbuf, vg), tdm_g1(TSEQ, HD, 32, HD, TSEQ));
        }

        if (blk < nblk) {                           // wave-uniform causal gate
            const int s0 = blk * 32;
            const _Float16* Ksb = smem + (blk & 1) * 2048;          // [key][64]
            const _Float16* Vsb = smem + 4096 + (blk & 1) * 2048;   // [feat][32]

            // S = Q K^T for 32 keys: two 16x16 f32 frags, 2 WMMAs each.
            v8f S[2];
#pragma unroll
            for (int j = 0; j < 2; ++j) {
                v8f sacc = {};
#pragma unroll
                for (int f = 0; f < 2; ++f) {
                    const _Float16* kp =
                        Ksb + (j * 16 + lo) * HD + f * 32 + hi * 8;
                    const v8h k0v = *(const v8h*)kp;
                    const v8h k1v = *(const v8h*)(kp + 16);
                    const v16h kf = CAT16(k0v, k1v);
                    sacc = __builtin_amdgcn_wmma_f32_16x16x32_f16(
                        false, qa[f], false, kf, (short)0, sacc, false, false);
                }
                S[j] = sacc;
            }

            if (s0 + 31 > t0) {                     // diagonal: causal mask
#pragma unroll
                for (int j = 0; j < 2; ++j)
#pragma unroll
                    for (int r = 0; r < 8; ++r) {
                        const int s = s0 + j * 16 + lo;
                        const int t = t0 + r + 8 * hi;
                        if (s > t) S[j][r] = -1e30f;
                    }
            }

            // online softmax: per-row max / sum across 16 lanes
            float mnew[8], alpha[8];
#pragma unroll
            for (int r = 0; r < 8; ++r) {
                float v = fmaxf(S[0][r], S[1][r]);
#pragma unroll
                for (int off = 1; off < 16; off <<= 1)
                    v = fmaxf(v, __shfl_xor(v, off, 32));
                mnew[r]  = fmaxf(m[r], v);
                alpha[r] = __expf(m[r] - mnew[r]);
            }
            v8f P0, P1;
#pragma unroll
            for (int r = 0; r < 8; ++r) {
                P0[r] = __expf(S[0][r] - mnew[r]);
                P1[r] = __expf(S[1][r] - mnew[r]);
            }
#pragma unroll
            for (int r = 0; r < 8; ++r) {
                float s = P0[r] + P1[r];
#pragma unroll
                for (int off = 1; off < 16; off <<= 1)
                    s += __shfl_xor(s, off, 32);
                l[r] = l[r] * alpha[r] + s;
                m[r] = mnew[r];
            }
#pragma unroll
            for (int c = 0; c < 4; ++c)
#pragma unroll
                for (int r = 0; r < 8; ++r) o[c][r] *= alpha[r];

            // P: C-layout -> A-layout via wave-private LDS tile (16x32 f16)
            _Float16* pst = smem + 8192 + w * 512;
#pragma unroll
            for (int r = 0; r < 8; ++r) {
                const int row = r + 8 * hi;
                pst[row * 32 + lo]      = (_Float16)P0[r];
                pst[row * 32 + 16 + lo] = (_Float16)P1[r];
            }
            asm volatile("s_wait_dscnt 0" ::: "memory");
            v16h pa;
            {
                const _Float16* pr = pst + lo * 32 + kb8;
                const v8h p0 = *(const v8h*)pr;
                const v8h p1 = *(const v8h*)(pr + 16);
                pa = CAT16(p0, p1);
            }
            asm volatile("" ::: "memory");

            // O += P V   (V tile staged transposed: [feat][32 keys])
#pragma unroll
            for (int c = 0; c < 4; ++c) {
                const _Float16* vp = Vsb + (c * 16 + lo) * 32 + hi * 8;
                const v8h v0 = *(const v8h*)vp;
                const v8h v1 = *(const v8h*)(vp + 16);
                const v16h vf = CAT16(v0, v1);
                o[c] = __builtin_amdgcn_wmma_f32_16x16x32_f16(
                    false, pa, false, vf, (short)0, o[c], false, false);
            }
        }
        __syncthreads();                // everyone done with buffers[blk&1]
    }

    // epilogue: normalize by l, scatter into [B, T, H*HD] fp32
#pragma unroll
    for (int c = 0; c < 4; ++c)
#pragma unroll
        for (int r = 0; r < 8; ++r) {
            const int row = t0 + r + 8 * hi;
            const int col = h * HD + c * 16 + lo;
            out[(size_t)(b * TSEQ + row) * DMODEL + col] = o[c][r] / l[r];
        }
}

// ---------------------------------------------------------------------------
extern "C" void kernel_launch(void* const* d_in, const int* in_sizes, int n_in,
                              void* d_out, int out_size, void* d_ws, size_t ws_size,
                              hipStream_t stream) {
    const float* x  = (const float*)d_in[0];
    const float* Wq = (const float*)d_in[1];
    const float* bq = (const float*)d_in[2];
    const float* Wk = (const float*)d_in[3];
    const float* bk = (const float*)d_in[4];
    const float* Wv = (const float*)d_in[5];
    const float* bv = (const float*)d_in[6];
    float* out = (float*)d_out;

    const size_t perTensor = (size_t)NB * NH * TSEQ * HD;   // 4M f16
    _Float16* Qb = (_Float16*)d_ws;
    _Float16* Kb = Qb + perTensor;
    _Float16* Vt = Kb + perTensor;
    _Float16* xh = Vt + perTensor;                          // 4M f16
    _Float16* Wt = xh + (size_t)NB * TSEQ * DMODEL;         // 3M f16 (~38MB tot)

    prep_kernel<<<2048, 256, 0, stream>>>(x, Wq, Wk, Wv, xh, Wt);
    qkv_proj_kernel<<<1536, 256, 0, stream>>>(xh, Wt, bq, bk, bv, Qb, Kb, Vt);
    attn_kernel<<<512, 256, 0, stream>>>(Qb, Kb, Vt, out);
}
